// Net_74010876444835
// MI455X (gfx1250) — matrix operand order repaired and checked by reading, not compiled, not run on hardware
//
#include <hip/hip_runtime.h>

// ---------------- problem constants ----------------
#define NPTS  32768
#define MCEN  8192
#define KNN   32
#define NEDGE (MCEN * KNN)
#define C1    128
#define C2    256
#define CEXP  1024
#define BNEPS 1e-5f

// ---------------- bf16 helpers (bit-level, RNE) ----------------
typedef __attribute__((ext_vector_type(16))) __bf16          v16bf;
typedef __attribute__((ext_vector_type(8)))  float           v8f;
typedef __attribute__((ext_vector_type(8)))  unsigned short  v8us;   // 16 bytes

__device__ __forceinline__ unsigned short f2bf_bits(float f) {
  unsigned int u = __float_as_uint(f);
  unsigned int r = (u + 0x7FFFu + ((u >> 16) & 1u)) >> 16;
  return (unsigned short)r;
}

union Frag16 { v16bf v; v8us h[2]; unsigned short u[16]; };

// SiLU with hardware reciprocal (v_exp_f32 + v_rcp_f32); BN downstream makes
// the ~1ulp rcp error irrelevant, and it avoids the IEEE div expansion.
__device__ __forceinline__ float siluf(float x) {
  return x * __builtin_amdgcn_rcpf(1.0f + __expf(-x));
}

// A-matrix fragment (16 rows x 32 K, bf16) from row-major LDS [rows][K],
// per CDNA5 ISA 7.12.2: lanes 0-15: M=lane, K={0..7,16..23};
// lanes 16-31: M=lane-16, K={8..15,24..31}.
// Each lane: two contiguous 16-byte runs -> 2x ds_load_b128.
__device__ __forceinline__ v16bf frag_a_lds(const unsigned short* p, int row0,
                                            int k0, int ld) {
  int lane = threadIdx.x & 31;
  int hs   = lane >> 4;
  int m    = row0 + (lane & 15);
  const v8us* q = (const v8us*)(p + m * ld + k0 + hs * 8);
  Frag16 f;
  f.h[0] = q[0];   // K = k0 + hs*8 .. +7
  f.h[1] = q[2];   // K = k0 + 16 + hs*8 .. +7
  return f.v;
}

// B-matrix fragment (32 K x 16 N, bf16) from TRANSPOSED LDS [N][K] (ld = K pitch):
// lanes 0-15: N=lane, K={0..7,16..23}; lanes 16-31: N=lane-16, K={8..15,24..31}.
// Each lane: two contiguous 16-byte runs -> 2x ds_load_b128.
__device__ __forceinline__ v16bf frag_b_ldsT(const unsigned short* p, int k0,
                                             int col0, int ld) {
  int lane = threadIdx.x & 31;
  int hs   = lane >> 4;
  int n    = col0 + (lane & 15);
  const v8us* q = (const v8us*)(p + n * ld + k0 + hs * 8);
  Frag16 f;
  f.h[0] = q[0];
  f.h[1] = q[2];
  return f.v;
}

__device__ __forceinline__ v8f wmma_bf16(v16bf a, v16bf b, v8f c) {
  return __builtin_amdgcn_wmma_f32_16x16x32_bf16(false, a, false, b,
                                                 (short)0, c, false, false);
}

// ---------------- kernel 1: moments of d = pos[j]-pos[c] over all edges ----
__launch_bounds__(256)
__global__ void k_dstats(const float* __restrict__ pos4,
                         const int* __restrict__ cidx,
                         const int* __restrict__ kidx,
                         float* __restrict__ dsum,   // [4]
                         float* __restrict__ dsq) {  // [10] upper-tri
  __shared__ float ls[16];
  if (threadIdx.x < 16) ls[threadIdx.x] = 0.f;
  __syncthreads();
  float s[4] = {0, 0, 0, 0};
  float q[10] = {0, 0, 0, 0, 0, 0, 0, 0, 0, 0};
  int stride = gridDim.x * blockDim.x;
  for (int e = blockIdx.x * blockDim.x + threadIdx.x; e < NEDGE; e += stride) {
    int m = e >> 5;
    int c = cidx[m];
    int j = kidx[e];
    float4 pj = *(const float4*)&pos4[j * 4];
    float4 pc = *(const float4*)&pos4[c * 4];
    float d[4] = {pj.x - pc.x, pj.y - pc.y, pj.z - pc.z, pj.w - pc.w};
#pragma unroll
    for (int x = 0; x < 4; ++x) s[x] += d[x];
    int qi = 0;
#pragma unroll
    for (int i = 0; i < 4; ++i)
#pragma unroll
      for (int jj = i; jj < 4; ++jj) q[qi++] += d[i] * d[jj];
  }
#pragma unroll
  for (int x = 0; x < 4; ++x) atomicAdd(&ls[x], s[x]);
#pragma unroll
  for (int x = 0; x < 10; ++x) atomicAdd(&ls[4 + x], q[x]);
  __syncthreads();
  if (threadIdx.x < 4)  atomicAdd(&dsum[threadIdx.x], ls[threadIdx.x]);
  if (threadIdx.x < 10) atomicAdd(&dsq[threadIdx.x], ls[4 + threadIdx.x]);
}

// -------- kernel 2: analytic BN1 (z1 = d@W1; bias cancels in BN) ----------
__launch_bounds__(128)
__global__ void k_fin_bn1(const float* __restrict__ dsum,
                          const float* __restrict__ dsq,
                          const float* __restrict__ W1,
                          const float* __restrict__ g1,
                          const float* __restrict__ be1,
                          float* __restrict__ s1, float* __restrict__ t1) {
  int c = threadIdx.x;
  const float invn = 1.0f / (float)NEDGE;
  float md[4];
#pragma unroll
  for (int j = 0; j < 4; ++j) md[j] = dsum[j] * invn;
  float E[4][4];
  int qi = 0;
#pragma unroll
  for (int i = 0; i < 4; ++i)
#pragma unroll
    for (int j = i; j < 4; ++j) {
      float v = dsq[qi++] * invn;
      E[i][j] = v; E[j][i] = v;
    }
  float w[4];
#pragma unroll
  for (int j = 0; j < 4; ++j) w[j] = W1[j * C1 + c];
  float mean = 0.f, var = 0.f;
#pragma unroll
  for (int j = 0; j < 4; ++j) mean += md[j] * w[j];
#pragma unroll
  for (int i = 0; i < 4; ++i)
#pragma unroll
    for (int j = 0; j < 4; ++j) var += w[i] * w[j] * (E[i][j] - md[i] * md[j]);
  var = fmaxf(var, 0.f);
  float S = g1[c] * rsqrtf(var + BNEPS);
  s1[c] = S;
  t1[c] = be1[c] - mean * S;
}

// -------- edge MLP kernel: h1 -> (h1 @ W2) via WMMA ----------------------
// mode 0: accumulate per-channel sum/sumsq of z2 (for BN2 stats)
// mode 1: apply BN2+SiLU and max-reduce over the 32 edges of each centroid
__launch_bounds__(256)
__global__ void k_edge_mlp(const float* __restrict__ pos4,
                           const int* __restrict__ cidx,
                           const int* __restrict__ kidx,
                           const float* __restrict__ W1,
                           const float* __restrict__ W2,
                           const float* __restrict__ s1,
                           const float* __restrict__ t1,
                           const float* __restrict__ s2,
                           const float* __restrict__ t2,
                           float* __restrict__ z2sum, float* __restrict__ z2sq,
                           float* __restrict__ xf,
                           unsigned short* __restrict__ xbf,
                           int mode) {
  __shared__ unsigned short sH1[128 * C1];   // 32 KB: h1 tile, row-major [edge][k]
  __shared__ unsigned short sBt[16 * C1];    //  4 KB: W2 chan-tile, transposed [n][k]
  __shared__ float sW1[4 * C1];              //  2 KB
  __shared__ float sC[128 * 16];             //  8 KB: f32 result staging
  __shared__ float lsum[C2], lsq[C2];        //  2 KB

  const int tid = threadIdx.x;
  for (int i = tid; i < 4 * C1; i += 256) sW1[i] = W1[i];
  if (mode == 0) { lsum[tid] = 0.f; lsq[tid] = 0.f; }
  __syncthreads();

  const int e0 = blockIdx.x * 128;   // 128 contiguous edges = 4 centroids
  {
    // build h1[128][128] in LDS: 2 threads per edge, 64 channels each
    int eL = tid >> 1;
    int ch0 = (tid & 1) * 64;
    int e = e0 + eL;
    int m = e >> 5;
    int c = cidx[m];
    int j = kidx[e];
    float4 pj = *(const float4*)&pos4[j * 4];
    float4 pc = *(const float4*)&pos4[c * 4];
    float d0 = pj.x - pc.x, d1 = pj.y - pc.y, d2 = pj.z - pc.z, d3 = pj.w - pc.w;
    for (int cc = 0; cc < 64; cc += 2) {
      int ch = ch0 + cc;
      float za = d0 * sW1[ch] + d1 * sW1[C1 + ch] +
                 d2 * sW1[2 * C1 + ch] + d3 * sW1[3 * C1 + ch];
      float zb = d0 * sW1[ch + 1] + d1 * sW1[C1 + ch + 1] +
                 d2 * sW1[2 * C1 + ch + 1] + d3 * sW1[3 * C1 + ch + 1];
      za = siluf(za * s1[ch] + t1[ch]);
      zb = siluf(zb * s1[ch + 1] + t1[ch + 1]);
      unsigned int pk = (unsigned int)f2bf_bits(za) |
                        ((unsigned int)f2bf_bits(zb) << 16);
      *(unsigned int*)&sH1[eL * C1 + ch] = pk;
    }
  }
  __syncthreads();

  const int wave = tid >> 5;
  v16bf a[4];
#pragma unroll
  for (int kc = 0; kc < 4; ++kc) a[kc] = frag_a_lds(sH1, wave * 16, kc * 32, C1);

  for (int t = 0; t < 16; ++t) {          // 16 channel tiles of 16 -> 256
    // stage W2[:,t*16..t*16+15] transposed: sBt[n][k], coalesced float4 reads
    for (int i = tid; i < 512; i += 256) {          // 128 k-rows x 4 col-chunks
      int r = i >> 2, c4 = (i & 3) * 4;
      float4 wv = *(const float4*)&W2[r * C2 + t * 16 + c4];
      sBt[(c4 + 0) * C1 + r] = f2bf_bits(wv.x);
      sBt[(c4 + 1) * C1 + r] = f2bf_bits(wv.y);
      sBt[(c4 + 2) * C1 + r] = f2bf_bits(wv.z);
      sBt[(c4 + 3) * C1 + r] = f2bf_bits(wv.w);
    }
    __syncthreads();
    v8f acc = {0.f, 0.f, 0.f, 0.f, 0.f, 0.f, 0.f, 0.f};
#pragma unroll
    for (int kc = 0; kc < 4; ++kc) {
      v16bf b = frag_b_ldsT(sBt, kc * 32, 0, C1);
      acc = wmma_bf16(a[kc], b, acc);
    }
    {
      int lane = tid & 31, hl = lane >> 4, col = lane & 15;
#pragma unroll
      for (int r = 0; r < 8; ++r)
        sC[(wave * 16 + hl * 8 + r) * 16 + col] = acc[r];
    }
    __syncthreads();
    if (mode == 0) {
      int colx = tid & 15, seg = tid >> 4;
      float bs = 0.f, bq = 0.f;
      for (int r = 0; r < 8; ++r) {
        float z = sC[(seg * 8 + r) * 16 + colx];
        bs += z; bq += z * z;
      }
      atomicAdd(&lsum[t * 16 + colx], bs);
      atomicAdd(&lsq[t * 16 + colx], bq);
    } else if (tid < 64) {
      int cen = tid >> 4, colx = tid & 15;
      int ch = t * 16 + colx;
      float ss = s2[ch], tt = t2[ch];
      float mx = -3.4e38f;
      for (int r = 0; r < 32; ++r) {
        float z = siluf(sC[(cen * 32 + r) * 16 + colx] * ss + tt);
        mx = fmaxf(mx, z);
      }
      int mg = blockIdx.x * 4 + cen;
      xf[mg * C2 + ch] = mx;
      xbf[mg * C2 + ch] = f2bf_bits(mx);
    }
    __syncthreads();
  }
  if (mode == 0) {
    atomicAdd(&z2sum[tid], lsum[tid]);
    atomicAdd(&z2sq[tid], lsq[tid]);
  }
}

// -------- generic bf16 WMMA GEMM with fused channel-stats ----------------
// C[row0..+127, col0..+127] = Xbf16 @ Wf32 ; per-channel sum/sumsq
__launch_bounds__(256)
__global__ void k_gemm_stats(const unsigned short* __restrict__ X,
                             const float* __restrict__ W,
                             float* __restrict__ Z,
                             float* __restrict__ gsum,
                             float* __restrict__ gsq,
                             int Kd, int Nout) {
  __shared__ unsigned short sA[128 * 32];   // 8 KB, row-major [row][k]
  __shared__ unsigned short sBt[128 * 32];  // 8 KB, transposed [n][k]
  __shared__ float lsum[128], lsq[128];
  const int tid  = threadIdx.x;
  const int row0 = blockIdx.x * 128;
  const int col0 = blockIdx.y * 128;
  const int wave = tid >> 5;

  v8f acc[8];
#pragma unroll
  for (int t = 0; t < 8; ++t) acc[t] = (v8f){0.f,0.f,0.f,0.f,0.f,0.f,0.f,0.f};
  if (tid < 128) { lsum[tid] = 0.f; lsq[tid] = 0.f; }

  for (int kk = 0; kk < Kd; kk += 32) {
    __syncthreads();
    // A: 128x32 bf16, 16-byte vector copies (coalesced)
    for (int i = tid; i < 512; i += 256) {
      int r = i >> 2, part = (i & 3) * 8;
      *(v8us*)&sA[r * 32 + part] =
          *(const v8us*)&X[(size_t)(row0 + r) * Kd + kk + part];
    }
    // B: 32x128 f32 -> bf16, transposed into [n][k]; coalesced float4 reads
    for (int i = tid; i < 1024; i += 256) {         // 32 k-rows x 32 col-chunks
      int r = i >> 5, c4 = (i & 31) * 4;
      float4 wv = *(const float4*)&W[(size_t)(kk + r) * Nout + col0 + c4];
      sBt[(c4 + 0) * 32 + r] = f2bf_bits(wv.x);
      sBt[(c4 + 1) * 32 + r] = f2bf_bits(wv.y);
      sBt[(c4 + 2) * 32 + r] = f2bf_bits(wv.z);
      sBt[(c4 + 3) * 32 + r] = f2bf_bits(wv.w);
    }
    __syncthreads();
    v16bf a = frag_a_lds(sA, wave * 16, 0, 32);
#pragma unroll
    for (int t = 0; t < 8; ++t) {
      v16bf b = frag_b_ldsT(sBt, 0, t * 16, 32);
      acc[t] = wmma_bf16(a, b, acc[t]);
    }
  }
  __syncthreads();

  int lane = tid & 31, hl = lane >> 4, col = lane & 15;
#pragma unroll
  for (int t = 0; t < 8; ++t) {
    float bs = 0.f, bq = 0.f;
    int ch = col0 + t * 16 + col;
#pragma unroll
    for (int r = 0; r < 8; ++r) {
      int row = row0 + wave * 16 + hl * 8 + r;
      float z = acc[t][r];
      Z[(size_t)row * Nout + ch] = z;
      bs += z; bq += z * z;
    }
    atomicAdd(&lsum[t * 16 + col], bs);
    atomicAdd(&lsq[t * 16 + col], bq);
  }
  __syncthreads();
  if (tid < 128) {
    atomicAdd(&gsum[col0 + tid], lsum[tid]);
    atomicAdd(&gsq[col0 + tid], lsq[tid]);
  }
}

// -------- BN finalize: fold (optional affine alpha) + BN into scale/shift --
// out = bn(alpha*h + beta) == h*sOut + tOut  (beta cancels inside BN)
__launch_bounds__(256)
__global__ void k_fin(const float* __restrict__ sum, const float* __restrict__ sq,
                      float invn,
                      const float* __restrict__ g, const float* __restrict__ be,
                      const float* __restrict__ alpha,
                      float* __restrict__ sOut, float* __restrict__ tOut, int C) {
  int c = blockIdx.x * blockDim.x + threadIdx.x;
  if (c >= C) return;
  float mh = sum[c] * invn;
  float vh = fmaxf(sq[c] * invn - mh * mh, 0.f);
  float a  = alpha ? alpha[c] : 1.f;
  float r  = rsqrtf(a * a * vh + BNEPS);
  float S  = g[c] * r;
  sOut[c] = S * a;
  tOut[c] = be[c] - a * mh * S;
}

// -------- elementwise: y = [silu](x*s+t [+res]); optional f32/bf16 out+stats
__launch_bounds__(256)
__global__ void k_elem(const float* __restrict__ X,
                       const float* __restrict__ s, const float* __restrict__ t,
                       const float* __restrict__ resid,
                       float* __restrict__ outF, unsigned short* __restrict__ outB,
                       float* __restrict__ gsum, float* __restrict__ gsq,
                       int n, int C, int do_silu) {
  extern __shared__ float sh[];
  float* lsum = sh;
  float* lsq  = sh + C;
  if (gsum) {
    for (int i = threadIdx.x; i < C; i += blockDim.x) { lsum[i] = 0.f; lsq[i] = 0.f; }
    __syncthreads();
  }
  size_t total = (size_t)n * C;
  size_t stride = (size_t)gridDim.x * blockDim.x;
  for (size_t idx = (size_t)blockIdx.x * blockDim.x + threadIdx.x; idx < total;
       idx += stride) {
    int c = (int)(idx & (size_t)(C - 1));
    float v = X[idx] * s[c] + t[c];
    if (resid) v += resid[idx];
    if (do_silu) v = siluf(v);
    if (outF) outF[idx] = v;
    if (outB) outB[idx] = f2bf_bits(v);
    if (gsum) { atomicAdd(&lsum[c], v); atomicAdd(&lsq[c], v * v); }
  }
  if (gsum) {
    __syncthreads();
    for (int i = threadIdx.x; i < C; i += blockDim.x) {
      atomicAdd(&gsum[i], lsum[i]);
      atomicAdd(&gsq[i], lsq[i]);
    }
  }
}

// ---------------------------------------------------------------------------
extern "C" void kernel_launch(void* const* d_in, const int* in_sizes, int n_in,
                              void* d_out, int out_size, void* d_ws, size_t ws_size,
                              hipStream_t stream) {
  (void)in_sizes; (void)n_in; (void)out_size; (void)ws_size;
  const float* pos4 = (const float*)d_in[0];
  const int*   cidx = (const int*)d_in[1];
  const int*   kidx = (const int*)d_in[2];
  int p = 3;
  const float* W1   = (const float*)d_in[p++];
  p++; /* b1 cancels in BN */
  const float* g1   = (const float*)d_in[p++];
  const float* be1  = (const float*)d_in[p++];
  const float* W2   = (const float*)d_in[p++];
  p++; /* b2 */
  const float* g2   = (const float*)d_in[p++];
  const float* be2  = (const float*)d_in[p++];
  const float* exW  = (const float*)d_in[p++];
  p++; /* exb */
  const float* exg  = (const float*)d_in[p++];
  const float* exbe = (const float*)d_in[p++];
  const float* dw1s = (const float*)d_in[p++];
  p++; /* dw1b */
  const float* dw1g = (const float*)d_in[p++];
  const float* dw1be= (const float*)d_in[p++];
  const float* pw1W = (const float*)d_in[p++];
  p++; /* pw1b */
  const float* pw1g = (const float*)d_in[p++];
  const float* pw1be= (const float*)d_in[p++];
  const float* cb1g = (const float*)d_in[p++];
  const float* cb1be= (const float*)d_in[p++];
  const float* dw2s = (const float*)d_in[p++];
  p++; /* dw2b */
  const float* dw2g = (const float*)d_in[p++];
  const float* dw2be= (const float*)d_in[p++];
  const float* pw2W = (const float*)d_in[p++];
  p++; /* pw2b */
  const float* pw2g = (const float*)d_in[p++];
  const float* pw2be= (const float*)d_in[p++];
  const float* cb2g = (const float*)d_in[p++];
  const float* cb2be= (const float*)d_in[p++];
  const float* prW  = (const float*)d_in[p++];
  p++; /* prb */
  const float* prg  = (const float*)d_in[p++];
  const float* prbe = (const float*)d_in[p++];

  // ---------------- workspace layout (all section starts 16B aligned) ------
  float* w = (float*)d_ws;
  size_t off = 0;
  float* dsum  = w + off; off += 4;
  float* dsq   = w + off; off += 12;
  float* z2sum = w + off; off += C2;
  float* z2sq  = w + off; off += C2;
  float* stS[8]; float* stQ[8];
  for (int i = 0; i < 8; ++i) { stS[i] = w + off; off += CEXP; stQ[i] = w + off; off += CEXP; }
  size_t zeroFloats = off;                      // stats region zeroed each call
  float* s1 = w + off; off += C1;  float* t1 = w + off; off += C1;
  float* s2 = w + off; off += C2;  float* t2 = w + off; off += C2;
  float* sE[8]; float* tE[8];
  for (int i = 0; i < 8; ++i) { sE[i] = w + off; off += CEXP; tE[i] = w + off; off += CEXP; }
  float* xf   = w + off; off += (size_t)MCEN * C2;
  float* bufZ = w + off; off += (size_t)MCEN * CEXP;
  float* bufH = w + off; off += (size_t)MCEN * CEXP;
  unsigned short* xbf    = (unsigned short*)(w + off); off += (size_t)MCEN * C2 / 2;
  unsigned short* bufXbf = (unsigned short*)(w + off); off += (size_t)MCEN * CEXP / 2;

  hipMemsetAsync(d_ws, 0, zeroFloats * sizeof(float), stream);

  const float invM = 1.0f / (float)MCEN;
  const size_t shmem = 2 * CEXP * sizeof(float);

  // ---- stage 1: edge MLP + max aggregation ----
  k_dstats<<<512, 256, 0, stream>>>(pos4, cidx, kidx, dsum, dsq);
  k_fin_bn1<<<1, 128, 0, stream>>>(dsum, dsq, W1, g1, be1, s1, t1);
  k_edge_mlp<<<NEDGE / 128, 256, 0, stream>>>(pos4, cidx, kidx, W1, W2, s1, t1,
                                              s2, t2, z2sum, z2sq, xf, xbf, 0);
  k_fin<<<1, 256, 0, stream>>>(z2sum, z2sq, 1.0f / (float)NEDGE, g2, be2,
                               nullptr, s2, t2, C2);
  k_edge_mlp<<<NEDGE / 128, 256, 0, stream>>>(pos4, cidx, kidx, W1, W2, s1, t1,
                                              s2, t2, z2sum, z2sq, xf, xbf, 1);

  // ---- stage 2: inverted residual block (per-point 1x1 convs) ----
  // expand 256 -> 1024
  k_gemm_stats<<<dim3(MCEN / 128, CEXP / 128), 256, 0, stream>>>(
      xbf, exW, bufZ, stS[0], stQ[0], C2, CEXP);
  k_fin<<<4, 256, 0, stream>>>(stS[0], stQ[0], invM, exg, exbe, nullptr, sE[0], tE[0], CEXP);
  k_elem<<<2048, 256, shmem, stream>>>(bufZ, sE[0], tE[0], nullptr, bufH, nullptr,
                                       stS[1], stQ[1], MCEN, CEXP, 1);
  // depthwise1 (affine) + BN + SiLU, emit bf16 for pw1
  k_fin<<<4, 256, 0, stream>>>(stS[1], stQ[1], invM, dw1g, dw1be, dw1s, sE[1], tE[1], CEXP);
  k_elem<<<2048, 256, shmem, stream>>>(bufH, sE[1], tE[1], nullptr, nullptr, bufXbf,
                                       nullptr, nullptr, MCEN, CEXP, 1);
  // pointwise1 1024 -> 1024
  k_gemm_stats<<<dim3(MCEN / 128, CEXP / 128), 256, 0, stream>>>(
      bufXbf, pw1W, bufZ, stS[2], stQ[2], CEXP, CEXP);
  k_fin<<<4, 256, 0, stream>>>(stS[2], stQ[2], invM, pw1g, pw1be, nullptr, sE[2], tE[2], CEXP);
  k_elem<<<2048, 256, shmem, stream>>>(bufZ, sE[2], tE[2], nullptr, bufH, nullptr,
                                       stS[3], stQ[3], MCEN, CEXP, 1);
  // cb1 BN + SiLU
  k_fin<<<4, 256, 0, stream>>>(stS[3], stQ[3], invM, cb1g, cb1be, nullptr, sE[3], tE[3], CEXP);
  k_elem<<<2048, 256, shmem, stream>>>(bufH, sE[3], tE[3], nullptr, bufZ, nullptr,
                                       stS[4], stQ[4], MCEN, CEXP, 1);
  // depthwise2 + BN + SiLU, emit bf16 for pw2
  k_fin<<<4, 256, 0, stream>>>(stS[4], stQ[4], invM, dw2g, dw2be, dw2s, sE[4], tE[4], CEXP);
  k_elem<<<2048, 256, shmem, stream>>>(bufZ, sE[4], tE[4], nullptr, nullptr, bufXbf,
                                       nullptr, nullptr, MCEN, CEXP, 1);
  // pointwise2 1024 -> 1024
  k_gemm_stats<<<dim3(MCEN / 128, CEXP / 128), 256, 0, stream>>>(
      bufXbf, pw2W, bufH, stS[5], stQ[5], CEXP, CEXP);
  k_fin<<<4, 256, 0, stream>>>(stS[5], stQ[5], invM, pw2g, pw2be, nullptr, sE[5], tE[5], CEXP);
  k_elem<<<2048, 256, shmem, stream>>>(bufH, sE[5], tE[5], nullptr, bufZ, nullptr,
                                       stS[6], stQ[6], MCEN, CEXP, 1);
  // cb2 BN (no act), emit bf16 for projection
  k_fin<<<4, 256, 0, stream>>>(stS[6], stQ[6], invM, cb2g, cb2be, nullptr, sE[6], tE[6], CEXP);
  k_elem<<<2048, 256, shmem, stream>>>(bufZ, sE[6], tE[6], nullptr, nullptr, bufXbf,
                                       nullptr, nullptr, MCEN, CEXP, 0);
  // project 1024 -> 256
  k_gemm_stats<<<dim3(MCEN / 128, C2 / 128), 256, 0, stream>>>(
      bufXbf, prW, bufH, stS[7], stQ[7], CEXP, C2);
  k_fin<<<1, 256, 0, stream>>>(stS[7], stQ[7], invM, prg, prbe, nullptr, sE[7], tE[7], C2);
  // final: silu(bn(Z3) + residual x)
  k_elem<<<2048, 256, 2 * C2 * sizeof(float), stream>>>(
      bufH, sE[7], tE[7], xf, (float*)d_out, nullptr, nullptr, nullptr, MCEN, C2, 1);
}
